// ConversationS2S_62826781606476
// MI455X (gfx1250) — matrix operand order repaired and checked by reading, not compiled, not run on hardware
//
#include <hip/hip_runtime.h>
#include <hip/hip_bf16.h>

// ---------------- Types for WMMA ----------------
typedef __attribute__((ext_vector_type(16))) __bf16 v16bf;
typedef __attribute__((ext_vector_type(8)))  __bf16 v8bf;
typedef __attribute__((ext_vector_type(8)))  float  v8f;

// Pack two fp32 into packed bf16 (round-to-nearest, ties-away) using V_PERM_B32:
// add 0x8000 rounding bias, then gather the two high 16-bit halves in one perm.
__device__ __forceinline__ unsigned pack2bf(float lo, float hi) {
    unsigned ul = __float_as_uint(lo) + 0x8000u;
    unsigned uh = __float_as_uint(hi) + 0x8000u;
    // result bytes: {uh[3], uh[2], ul[3], ul[2]}  (selectors: b=ul -> 0..3, a=uh -> 4..7)
    return __builtin_amdgcn_perm(uh, ul, 0x07060302u);
}

// ---------------- Generic batched WMMA GEMM ----------------
// C[z][m][n] = sum_k A[z](m,k) * B[z](n,k)  (+bias[n])  (optional relu)
//   A(m,k) at A + offA(z) + m*a_rs + k           (k contiguous)
//   B(n,k) at B + offB(z) + n*b_ns + k*b_ks      (strided -> free transpose)
//   C(m,n) at C + offC(z) + m*c_rs + n
//   off*(z) = (z/bdiv)*bs0 + (z%bdiv)*bs1
// Block tile 128(M) x 64(N) x 32(K); 8 waves (4Mx2N); wave tile 32x32 = 4 WMMA accums.
// Register-double-buffered global->LDS staging with fp32->bf16 perm-pack.
__global__ __launch_bounds__(256) void gemm_bf16_wmma(
    const float* __restrict__ A, long long a_rs, long long a_bs0, long long a_bs1,
    const float* __restrict__ B, long long b_ns, long long b_ks, long long b_bs0, long long b_bs1,
    float* __restrict__ C, long long c_rs, long long c_bs0, long long c_bs1,
    const float* __restrict__ bias, long long bias_bs,
    int K, int bdiv, int flags)
{
    __shared__ unsigned short sA[128 * 32];   // [m][k]
    __shared__ unsigned short sB[64 * 32];    // [n][k]

    const int z  = blockIdx.z;
    const int z0 = z / bdiv;
    const int z1 = z % bdiv;

    const float* Ab = A + z0 * a_bs0 + z1 * a_bs1 + (long long)blockIdx.y * 128 * a_rs;
    const float* Bb = B + z0 * b_bs0 + z1 * b_bs1 + (long long)blockIdx.x * 64 * b_ns;
    float*       Cb = C + z0 * c_bs0 + z1 * c_bs1 + (long long)blockIdx.y * 128 * c_rs
                        + (long long)blockIdx.x * 64;

    const int tid  = threadIdx.x;
    const int lane = tid & 31;
    const int wave = tid >> 5;
    const int wm   = wave & 3;      // 4 M-slabs of 32 rows
    const int wn   = wave >> 2;     // 2 N-slabs of 32 cols
    const int grp  = lane >> 4;     // K-half selector per ISA layout
    const int l16  = lane & 15;

    // staging coordinates (quad = 4 consecutive k elements)
    // A: 128x32 = 1024 quads -> 4 per thread;  B: 64x32 = 512 quads -> 2 per thread
    int arow[4], aqk[4];
    #pragma unroll
    for (int i = 0; i < 4; ++i) { int jq = tid + i * 256; arow[i] = jq >> 3; aqk[i] = (jq & 7) * 4; }
    int brow[2], bqk[2];
    #pragma unroll
    for (int i = 0; i < 2; ++i) { int jq = tid + i * 256; brow[i] = jq >> 3; bqk[i] = (jq & 7) * 4; }

    const bool b_contig = (b_ks == 1);

    float4 pa[4];
    float4 pb[2];

    auto fetch = [&](int kk) {
        #pragma unroll
        for (int i = 0; i < 4; ++i)
            pa[i] = *reinterpret_cast<const float4*>(Ab + (long long)arow[i] * a_rs + kk + aqk[i]);
        if (b_contig) {
            #pragma unroll
            for (int i = 0; i < 2; ++i)
                pb[i] = *reinterpret_cast<const float4*>(Bb + (long long)brow[i] * b_ns + kk + bqk[i]);
        } else {
            #pragma unroll
            for (int i = 0; i < 2; ++i) {
                const float* bp = Bb + (long long)brow[i] * b_ns + (long long)(kk + bqk[i]) * b_ks;
                pb[i].x = bp[0];
                pb[i].y = bp[b_ks];
                pb[i].z = bp[2 * b_ks];
                pb[i].w = bp[3 * b_ks];
            }
        }
    };

    v8f acc00 = {}, acc01 = {}, acc10 = {}, acc11 = {};

    fetch(0);
    for (int k0 = 0; k0 < K; k0 += 32) {
        // commit prefetched tile to LDS (bf16 perm-packed, 64-bit DS stores)
        #pragma unroll
        for (int i = 0; i < 4; ++i) {
            uint2 u;
            u.x = pack2bf(pa[i].x, pa[i].y);
            u.y = pack2bf(pa[i].z, pa[i].w);
            *reinterpret_cast<uint2*>(&sA[arow[i] * 32 + aqk[i]]) = u;
        }
        #pragma unroll
        for (int i = 0; i < 2; ++i) {
            uint2 u;
            u.x = pack2bf(pb[i].x, pb[i].y);
            u.y = pack2bf(pb[i].z, pb[i].w);
            *reinterpret_cast<uint2*>(&sB[brow[i] * 32 + bqk[i]]) = u;
        }
        __syncthreads();

        if (k0 + 32 < K) fetch(k0 + 32);   // overlap next-tile global loads with WMMAs

        // A fragments: lane (grp,l16): K = grp*8 + 0..7 and 16 + grp*8 + 0..7
        const int ar0 = (wm * 32 + l16) * 32 + grp * 8;
        const int ar1 = (wm * 32 + 16 + l16) * 32 + grp * 8;
        const v8bf a00 = *reinterpret_cast<const v8bf*>(&sA[ar0]);
        const v8bf a01 = *reinterpret_cast<const v8bf*>(&sA[ar0 + 16]);
        const v8bf a10 = *reinterpret_cast<const v8bf*>(&sA[ar1]);
        const v8bf a11 = *reinterpret_cast<const v8bf*>(&sA[ar1 + 16]);
        v16bf af0, af1;
        #pragma unroll
        for (int i = 0; i < 8; ++i) {
            af0[i] = a00[i]; af0[i + 8] = a01[i];
            af1[i] = a10[i]; af1[i + 8] = a11[i];
        }
        // B fragments: lane (grp,l16): N = l16 (+tile), K = grp*8 + 0..7 and 16 + grp*8 + 0..7
        const int br0 = (wn * 32 + l16) * 32 + grp * 8;
        const int br1 = (wn * 32 + 16 + l16) * 32 + grp * 8;
        const v8bf b00 = *reinterpret_cast<const v8bf*>(&sB[br0]);
        const v8bf b01 = *reinterpret_cast<const v8bf*>(&sB[br0 + 16]);
        const v8bf b10 = *reinterpret_cast<const v8bf*>(&sB[br1]);
        const v8bf b11 = *reinterpret_cast<const v8bf*>(&sB[br1 + 16]);
        v16bf bf0, bf1;
        #pragma unroll
        for (int i = 0; i < 8; ++i) {
            bf0[i] = b00[i]; bf0[i + 8] = b01[i];
            bf1[i] = b10[i]; bf1[i + 8] = b11[i];
        }

        acc00 = __builtin_amdgcn_wmma_f32_16x16x32_bf16(false, af0, false, bf0, (short)0, acc00, false, false);
        acc01 = __builtin_amdgcn_wmma_f32_16x16x32_bf16(false, af0, false, bf1, (short)0, acc01, false, false);
        acc10 = __builtin_amdgcn_wmma_f32_16x16x32_bf16(false, af1, false, bf0, (short)0, acc10, false, false);
        acc11 = __builtin_amdgcn_wmma_f32_16x16x32_bf16(false, af1, false, bf1, (short)0, acc11, false, false);

        __syncthreads();
    }

    // epilogue: C/D layout: N = l16, M = r + 8*grp per VGPR r
    const int n0 = wn * 32 + l16;
    const int n1 = n0 + 16;
    float bv0 = 0.0f, bv1 = 0.0f;
    if (flags & 1) {
        const float* biasb = bias + z0 * bias_bs + blockIdx.x * 64;
        bv0 = biasb[n0];
        bv1 = biasb[n1];
    }
    #pragma unroll
    for (int r = 0; r < 8; ++r) {
        const int m0 = wm * 32 + r + 8 * grp;
        const int m1 = m0 + 16;
        float v00 = acc00[r] + bv0, v01 = acc01[r] + bv1;
        float v10 = acc10[r] + bv0, v11 = acc11[r] + bv1;
        if (flags & 2) {
            v00 = fmaxf(v00, 0.0f); v01 = fmaxf(v01, 0.0f);
            v10 = fmaxf(v10, 0.0f); v11 = fmaxf(v11, 0.0f);
        }
        Cb[(long long)m0 * c_rs + n0] = v00;
        Cb[(long long)m0 * c_rs + n1] = v01;
        Cb[(long long)m1 * c_rs + n0] = v10;
        Cb[(long long)m1 * c_rs + n1] = v11;
    }
}

// ---------------- Embedding: concat per-quantizer token emb + positional ----------------
// codes [B,8,T] int32; tok [8,1024,128]; pos [MAXSEQ,1024]; out [B*T,1024]
__global__ __launch_bounds__(256) void embed_kernel(
    const int* __restrict__ codes, const float* __restrict__ tok,
    const float* __restrict__ pos, float* __restrict__ out, int T)
{
    const int bt = blockIdx.x;
    const int b = bt / T, t = bt % T;
    const int d0 = threadIdx.x * 4;
    const int q = d0 >> 7;              // quad never crosses 128-boundary
    const int code = codes[(b * 8 + q) * T + t];
    #pragma unroll
    for (int i = 0; i < 4; ++i) {
        int d = d0 + i;
        int e = d & 127;
        out[(long long)bt * 1024 + d] =
            tok[((long long)q * 1024 + code) * 128 + e] + pos[(long long)t * 1024 + d];
    }
}

// ---------------- Row softmax with scale + optional causal mask (Tk==512) ----------------
__global__ __launch_bounds__(256) void softmax_kernel(
    float* __restrict__ S, int Tq, int causal, float scale)
{
    __shared__ float red[256];
    const long long rowIdx = blockIdx.x;
    const int row = (int)(rowIdx % Tq);
    float* p = S + rowIdx * 512;
    const int tid = threadIdx.x;

    float v0 = p[tid] * scale;
    float v1 = p[tid + 256] * scale;
    if (causal) {
        if (tid > row)       v0 = -1e30f;
        if (tid + 256 > row) v1 = -1e30f;
    }
    red[tid] = fmaxf(v0, v1);
    __syncthreads();
    for (int s = 128; s > 0; s >>= 1) { if (tid < s) red[tid] = fmaxf(red[tid], red[tid + s]); __syncthreads(); }
    const float m = red[0];
    __syncthreads();
    v0 = __expf(v0 - m);
    v1 = __expf(v1 - m);
    red[tid] = v0 + v1;
    __syncthreads();
    for (int s = 128; s > 0; s >>= 1) { if (tid < s) red[tid] += red[tid + s]; __syncthreads(); }
    const float inv = 1.0f / red[0];
    p[tid]       = v0 * inv;
    p[tid + 256] = v1 * inv;
}

// ---------------- Residual + LayerNorm over D=1024 (out may alias res) ----------------
__global__ __launch_bounds__(256) void ln_residual_kernel(
    const float* __restrict__ x, const float* __restrict__ res,
    const float* __restrict__ g, const float* __restrict__ bta,
    float* __restrict__ out)
{
    __shared__ float red[256];
    const long long row = blockIdx.x;
    const float* xp = x + row * 1024;
    const float* rp = res + row * 1024;
    const int tid = threadIdx.x;

    float v[4];
    float s = 0.0f;
    #pragma unroll
    for (int i = 0; i < 4; ++i) { v[i] = xp[tid * 4 + i] + rp[tid * 4 + i]; s += v[i]; }
    red[tid] = s;
    __syncthreads();
    for (int st = 128; st > 0; st >>= 1) { if (tid < st) red[tid] += red[tid + st]; __syncthreads(); }
    const float mean = red[0] * (1.0f / 1024.0f);
    __syncthreads();
    s = 0.0f;
    #pragma unroll
    for (int i = 0; i < 4; ++i) { float d = v[i] - mean; s += d * d; }
    red[tid] = s;
    __syncthreads();
    for (int st = 128; st > 0; st >>= 1) { if (tid < st) red[tid] += red[tid + st]; __syncthreads(); }
    const float rstd = rsqrtf(red[0] * (1.0f / 1024.0f) + 1e-5f);
    float* op = out + row * 1024;
    #pragma unroll
    for (int i = 0; i < 4; ++i) {
        int d = tid * 4 + i;
        op[d] = (v[i] - mean) * rstd * g[d] + bta[d];
    }
}

// ---------------- Host-side orchestration ----------------
static void launch_gemm(hipStream_t s,
    const float* A, long long a_rs, long long a_bs0, long long a_bs1,
    const float* B, long long b_ns, long long b_ks, long long b_bs0, long long b_bs1,
    float* C, long long c_rs, long long c_bs0, long long c_bs1,
    const float* bias, long long bias_bs,
    int M, int N, int K, int Z, int bdiv, int flags)
{
    dim3 grid((unsigned)(N / 64), (unsigned)(M / 128), (unsigned)Z);
    gemm_bf16_wmma<<<grid, 256, 0, s>>>(A, a_rs, a_bs0, a_bs1,
                                        B, b_ns, b_ks, b_bs0, b_bs1,
                                        C, c_rs, c_bs0, c_bs1,
                                        bias, bias_bs, K, bdiv, flags);
}

extern "C" void kernel_launch(void* const* d_in, const int* in_sizes, int n_in,
                              void* d_out, int out_size, void* d_ws, size_t ws_size,
                              hipStream_t stream)
{
    (void)in_sizes; (void)n_in; (void)out_size; (void)ws_size;
    typedef long long ll;
    const int Bsz = 4, Qn = 8, T = 512, D = 1024, H = 16, HD = 64, FF = 4096, L = 4, E = 128, V = 1024;
    const int BT = Bsz * T;                 // 2048 tokens
    const float SCALE = 0.125f;             // 1/sqrt(64)

    // inputs (setup_inputs dict order)
    const int*   input_codes  = (const int*)  d_in[0];
    const int*   target_codes = (const int*)  d_in[1];
    const float* tok_emb      = (const float*)d_in[2];
    const float* pos_emb      = (const float*)d_in[3];
    const float* e_qkv_w  = (const float*)d_in[4];
    const float* e_out_w  = (const float*)d_in[5];
    const float* e_ff1_w  = (const float*)d_in[6];
    const float* e_ff2_w  = (const float*)d_in[7];
    const float* d_sqkv_w = (const float*)d_in[8];
    const float* d_sout_w = (const float*)d_in[9];
    const float* d_cqkv_w = (const float*)d_in[10];
    const float* d_cout_w = (const float*)d_in[11];
    const float* d_ff1_w  = (const float*)d_in[12];
    const float* d_ff2_w  = (const float*)d_in[13];
    const float* head_w   = (const float*)d_in[14];
    const float* e_qkv_b  = (const float*)d_in[15];
    const float* e_out_b  = (const float*)d_in[16];
    const float* e_ff1_b  = (const float*)d_in[17];
    const float* e_ff2_b  = (const float*)d_in[18];
    const float* d_sqkv_b = (const float*)d_in[19];
    const float* d_sout_b = (const float*)d_in[20];
    const float* d_cqkv_b = (const float*)d_in[21];
    const float* d_cout_b = (const float*)d_in[22];
    const float* d_ff1_b  = (const float*)d_in[23];
    const float* d_ff2_b  = (const float*)d_in[24];
    const float* head_b   = (const float*)d_in[25];
    const float* e_ln1_w = (const float*)d_in[26];
    const float* e_ln1_b = (const float*)d_in[27];
    const float* e_ln2_w = (const float*)d_in[28];
    const float* e_ln2_b = (const float*)d_in[29];
    const float* d_ln1_w = (const float*)d_in[30];
    const float* d_ln1_b = (const float*)d_in[31];
    const float* d_ln2_w = (const float*)d_in[32];
    const float* d_ln2_b = (const float*)d_in[33];
    const float* d_ln3_w = (const float*)d_in[34];
    const float* d_ln3_b = (const float*)d_in[35];

    float* out = (float*)d_out;

    // workspace layout (floats): ~120 MB total
    float* X    = (float*)d_ws;                 // [BT, D]  encoder state / memory
    float* Y    = X    + (ll)BT * D;            // [BT, D]  decoder state
    float* QKV  = Y    + (ll)BT * D;            // [BT, 3D] fused qkv (reused as Qc|KVc)
    float* SC   = QKV  + (ll)BT * 3 * D;        // [B*H, T, T] scores (reused as FFN hidden)
    float* ATTO = SC   + (ll)Bsz * H * T * T;   // [BT, D]
    float* PROJ = ATTO + (ll)BT * D;            // [BT, D]
    float* FFH  = SC;                           // [BT, FF] aliases scores
    float* Qc   = QKV;                          // [BT, D]
    float* KVc  = QKV + (ll)BT * D;             // [BT, 2D]

    const ll TT = (ll)T * T;

    // ===== Encoder =====
    embed_kernel<<<BT, 256, 0, stream>>>(input_codes, tok_emb, pos_emb, X, T);

    for (int l = 0; l < L; ++l) {
        const float* wqkv = e_qkv_w + (ll)l * 3 * D * D;
        const float* bqkv = e_qkv_b + (ll)l * 3 * D;
        // fused QKV projection
        launch_gemm(stream, X, D, 0, 0, wqkv, D, 1, 0, 0, QKV, 3 * D, 0, 0,
                    bqkv, 0, BT, 3 * D, D, 1, 1, 1);
        // scores = Q @ K^T  (batched over b,h)
        launch_gemm(stream, QKV, 3 * D, (ll)T * 3 * D, HD,
                    QKV + D, 3 * D, 1, (ll)T * 3 * D, HD,
                    SC, T, (ll)H * TT, TT, nullptr, 0,
                    T, T, HD, Bsz * H, H, 0);
        softmax_kernel<<<Bsz * H * T, 256, 0, stream>>>(SC, T, 0, SCALE);
        // O = P @ V
        launch_gemm(stream, SC, T, (ll)H * TT, TT,
                    QKV + 2 * D, 1, 3 * D, (ll)T * 3 * D, HD,
                    ATTO, D, (ll)T * D, HD, nullptr, 0,
                    T, HD, T, Bsz * H, H, 0);
        // out projection + residual LN
        launch_gemm(stream, ATTO, D, 0, 0, e_out_w + (ll)l * D * D, D, 1, 0, 0,
                    PROJ, D, 0, 0, e_out_b + (ll)l * D, 0, BT, D, D, 1, 1, 1);
        ln_residual_kernel<<<BT, 256, 0, stream>>>(PROJ, X, e_ln1_w + (ll)l * D, e_ln1_b + (ll)l * D, X);
        // FFN
        launch_gemm(stream, X, D, 0, 0, e_ff1_w + (ll)l * FF * D, D, 1, 0, 0,
                    FFH, FF, 0, 0, e_ff1_b + (ll)l * FF, 0, BT, FF, D, 1, 1, 3);
        launch_gemm(stream, FFH, FF, 0, 0, e_ff2_w + (ll)l * D * FF, FF, 1, 0, 0,
                    PROJ, D, 0, 0, e_ff2_b + (ll)l * D, 0, BT, D, FF, 1, 1, 1);
        ln_residual_kernel<<<BT, 256, 0, stream>>>(PROJ, X, e_ln2_w + (ll)l * D, e_ln2_b + (ll)l * D, X);
    }
    // X now holds encoder memory.

    // ===== Decoder =====
    embed_kernel<<<BT, 256, 0, stream>>>(target_codes, tok_emb, pos_emb, Y, T);

    for (int l = 0; l < L; ++l) {
        // --- causal self-attention ---
        launch_gemm(stream, Y, D, 0, 0, d_sqkv_w + (ll)l * 3 * D * D, D, 1, 0, 0,
                    QKV, 3 * D, 0, 0, d_sqkv_b + (ll)l * 3 * D, 0, BT, 3 * D, D, 1, 1, 1);
        launch_gemm(stream, QKV, 3 * D, (ll)T * 3 * D, HD,
                    QKV + D, 3 * D, 1, (ll)T * 3 * D, HD,
                    SC, T, (ll)H * TT, TT, nullptr, 0,
                    T, T, HD, Bsz * H, H, 0);
        softmax_kernel<<<Bsz * H * T, 256, 0, stream>>>(SC, T, 1, SCALE);
        launch_gemm(stream, SC, T, (ll)H * TT, TT,
                    QKV + 2 * D, 1, 3 * D, (ll)T * 3 * D, HD,
                    ATTO, D, (ll)T * D, HD, nullptr, 0,
                    T, HD, T, Bsz * H, H, 0);
        launch_gemm(stream, ATTO, D, 0, 0, d_sout_w + (ll)l * D * D, D, 1, 0, 0,
                    PROJ, D, 0, 0, d_sout_b + (ll)l * D, 0, BT, D, D, 1, 1, 1);
        ln_residual_kernel<<<BT, 256, 0, stream>>>(PROJ, Y, d_ln1_w + (ll)l * D, d_ln1_b + (ll)l * D, Y);

        // --- cross-attention: Q from Y, K/V from memory X ---
        const float* wc = d_cqkv_w + (ll)l * 3 * D * D;
        const float* bc = d_cqkv_b + (ll)l * 3 * D;
        launch_gemm(stream, Y, D, 0, 0, wc, D, 1, 0, 0,
                    Qc, D, 0, 0, bc, 0, BT, D, D, 1, 1, 1);
        launch_gemm(stream, X, D, 0, 0, wc + (ll)D * D, D, 1, 0, 0,
                    KVc, 2 * D, 0, 0, bc + D, 0, BT, 2 * D, D, 1, 1, 1);
        launch_gemm(stream, Qc, D, (ll)T * D, HD,
                    KVc, 2 * D, 1, (ll)T * 2 * D, HD,
                    SC, T, (ll)H * TT, TT, nullptr, 0,
                    T, T, HD, Bsz * H, H, 0);
        softmax_kernel<<<Bsz * H * T, 256, 0, stream>>>(SC, T, 0, SCALE);
        launch_gemm(stream, SC, T, (ll)H * TT, TT,
                    KVc + D, 1, 2 * D, (ll)T * 2 * D, HD,
                    ATTO, D, (ll)T * D, HD, nullptr, 0,
                    T, HD, T, Bsz * H, H, 0);
        launch_gemm(stream, ATTO, D, 0, 0, d_cout_w + (ll)l * D * D, D, 1, 0, 0,
                    PROJ, D, 0, 0, d_cout_b + (ll)l * D, 0, BT, D, D, 1, 1, 1);
        ln_residual_kernel<<<BT, 256, 0, stream>>>(PROJ, Y, d_ln2_w + (ll)l * D, d_ln2_b + (ll)l * D, Y);

        // --- FFN ---
        launch_gemm(stream, Y, D, 0, 0, d_ff1_w + (ll)l * FF * D, D, 1, 0, 0,
                    FFH, FF, 0, 0, d_ff1_b + (ll)l * FF, 0, BT, FF, D, 1, 1, 3);
        launch_gemm(stream, FFH, FF, 0, 0, d_ff2_w + (ll)l * D * FF, FF, 1, 0, 0,
                    PROJ, D, 0, 0, d_ff2_b + (ll)l * D, 0, BT, D, FF, 1, 1, 1);
        ln_residual_kernel<<<BT, 256, 0, stream>>>(PROJ, Y, d_ln3_w + (ll)l * D, d_ln3_b + (ll)l * D, Y);
    }

    // ===== Head: out[b,t,q,v] = Y[b,t,q*E:..] @ head_w[q]^T + head_b[q] =====
    launch_gemm(stream, Y, D, E, 0,
                head_w, E, 1, (ll)V * E, 0,
                out, (ll)Qn * V, (ll)V, 0,
                head_b, (ll)V,
                BT, V, E, Qn, 1, 1);
}
